// BlockSparseAttention_61392262529136
// MI455X (gfx1250) — matrix operand order repaired and checked by reading, not compiled, not run on hardware
//
#include <hip/hip_runtime.h>

// Block-sparse attention for MI455X (gfx1250, wave32, WMMA).
// B=2, S=2048, D=1024, H=16, DH=64, NB=16, BLK=128.
// Compute-bound -> all matmuls on v_wmma_f32_16x16x32_f16. Attention K/V
// tiles double-buffered in LDS via GLOBAL_LOAD_ASYNC_TO_LDS_B128: next kept
// block's copies fly while current block's 32 WMMAs execute; S_WAIT_ASYNCCNT 8
// releases only the older (current) copy group (async loads complete in order).

#define B_   2
#define S_   2048
#define D_   1024
#define H_   16
#define DH_  64
#define NB_  16
#define BLK_ 128

typedef _Float16 half_t;
typedef __attribute__((ext_vector_type(16))) _Float16 v16h;
typedef __attribute__((ext_vector_type(8)))  _Float16 v8h;
typedef __attribute__((ext_vector_type(8)))  float    v8f;
typedef int v4i __attribute__((vector_size(16)));
typedef __attribute__((address_space(1))) v4i as1_v4i;
typedef __attribute__((address_space(3))) v4i as3_v4i;

#if defined(__has_builtin)
#if __has_builtin(__builtin_amdgcn_global_load_async_to_lds_b128) && \
    __has_builtin(__builtin_amdgcn_s_wait_asynccnt)
#define USE_ASYNC_LDS 1
#endif
#endif

__device__ __forceinline__ void copy16_g2l(half_t* lds, const half_t* g) {
#ifdef USE_ASYNC_LDS
  __builtin_amdgcn_global_load_async_to_lds_b128(
      (as1_v4i*)(uintptr_t)g, (as3_v4i*)lds, 0, 0);
#else
  *reinterpret_cast<v8h*>(lds) = *reinterpret_cast<const v8h*>(g);
#endif
}

// wait until at most `n` async copies remain outstanding (older ones done)
__device__ __forceinline__ void async_wait_keep8() {
#ifdef USE_ASYNC_LDS
  __builtin_amdgcn_s_wait_asynccnt(8);
#endif
}
__device__ __forceinline__ void async_wait_all() {
#ifdef USE_ASYNC_LDS
  __builtin_amdgcn_s_wait_asynccnt(0);
#endif
}

// ---------------------------------------------------------------------------
__device__ __forceinline__ v8f wmma_f16(v16h a, v16h b, v8f c) {
  return __builtin_amdgcn_wmma_f32_16x16x32_f16(false, a, false, b,
                                                (short)0, c, false, false);
}

// A-matrix 16x32 f16 fragment from row-major [*, lda] memory.
__device__ __forceinline__ v16h load_a16x32(const half_t* __restrict__ A,
                                            int lda, int row0, int koff) {
  const int lane = threadIdx.x & 31;
  const int m    = row0 + (lane & 15);
  const int kb   = koff + ((lane >> 4) << 3);
  v8h lo = *reinterpret_cast<const v8h*>(A + (size_t)m * lda + kb);
  v8h hi = *reinterpret_cast<const v8h*>(A + (size_t)m * lda + kb + 16);
  v16h r;
#pragma unroll
  for (int i = 0; i < 8; ++i) { r[i] = lo[i]; r[8 + i] = hi[i]; }
  return r;
}

// B-matrix 32x16 f16 fragment; Bm[n*ldb + k], K contiguous per column n.
__device__ __forceinline__ v16h load_b32x16(const half_t* __restrict__ Bm,
                                            int ldb, int n0, int koff) {
  const int lane = threadIdx.x & 31;
  const int n    = n0 + (lane & 15);
  const int kb   = koff + ((lane >> 4) << 3);
  v8h lo = *reinterpret_cast<const v8h*>(Bm + (size_t)n * ldb + kb);
  v8h hi = *reinterpret_cast<const v8h*>(Bm + (size_t)n * ldb + kb + 16);
  v16h r;
#pragma unroll
  for (int i = 0; i < 8; ++i) { r[i] = lo[i]; r[8 + i] = hi[i]; }
  return r;
}

// ---------------------------------------------------------------------------
__global__ void cvt_f32_to_f16(const float* __restrict__ src,
                               half_t* __restrict__ dst, int n) {
  int i = blockIdx.x * blockDim.x + threadIdx.x;
  if (i < n) dst[i] = (half_t)src[i];
}

// ---------------------------------------------------------------------------
// GEMM: C[4096,1024] = Xh @ Wh^T. k-loop unrolled x2 with two named fragment
// sets (no cross-set register moves). Wave tile 32x64; WG = 8 waves = 256x64.
// mode 0: f16 head-split [B,H,S,DH] | mode 1: f16 [B,H,DH,S] | mode 2: f32+bias
// ---------------------------------------------------------------------------
__global__ __launch_bounds__(256) void gemm_xwT(const half_t* __restrict__ Xh,
                                                const half_t* __restrict__ Wh,
                                                void* __restrict__ outp,
                                                const float* __restrict__ bias,
                                                int mode) {
  const int wave = threadIdx.x >> 5;
  const int lane = threadIdx.x & 31;
  const int row0 = blockIdx.x * 256 + wave * 32;  // M
  const int col0 = blockIdx.y * 64;               // N

  v8f acc0[4] = {}, acc1[4] = {};

  // set A: fragments for k = 0
  v16h a0A = load_a16x32(Xh, D_, row0, 0);
  v16h a1A = load_a16x32(Xh, D_, row0 + 16, 0);
  v16h bA[4];
#pragma unroll
  for (int j = 0; j < 4; ++j) bA[j] = load_b32x16(Wh, D_, col0 + 16 * j, 0);

  for (int k = 0; k < D_; k += 64) {   // 16 iterations, 16 WMMA each
    // load set B for k+32 (k+32 <= 992 always valid)
    __builtin_prefetch(Xh + (size_t)(row0 + (lane & 15)) * D_ + k + 64, 0, 1);
    v16h a0B = load_a16x32(Xh, D_, row0, k + 32);
    v16h a1B = load_a16x32(Xh, D_, row0 + 16, k + 32);
    v16h bB[4];
#pragma unroll
    for (int j = 0; j < 4; ++j) bB[j] = load_b32x16(Wh, D_, col0 + 16 * j, k + 32);

#pragma unroll
    for (int j = 0; j < 4; ++j) {
      acc0[j] = wmma_f16(a0A, bA[j], acc0[j]);
      acc1[j] = wmma_f16(a1A, bA[j], acc1[j]);
    }

    // reload set A for k+64 (clamped on last iteration; redundant but branchless)
    const int kn = (k + 64 < D_) ? (k + 64) : 0;
    a0A = load_a16x32(Xh, D_, row0, kn);
    a1A = load_a16x32(Xh, D_, row0 + 16, kn);
#pragma unroll
    for (int j = 0; j < 4; ++j) bA[j] = load_b32x16(Wh, D_, col0 + 16 * j, kn);

#pragma unroll
    for (int j = 0; j < 4; ++j) {
      acc0[j] = wmma_f16(a0B, bB[j], acc0[j]);
      acc1[j] = wmma_f16(a1B, bB[j], acc1[j]);
    }
  }

  const int rbase = (lane >> 4) << 3;
  const int cIn   = lane & 15;
#pragma unroll
  for (int hv = 0; hv < 2; ++hv) {
    const v8f* acc = hv ? acc1 : acc0;
#pragma unroll
    for (int j = 0; j < 4; ++j) {
      const int col = col0 + 16 * j + cIn;
      const int h = col >> 6, dh = col & 63;
#pragma unroll
      for (int r = 0; r < 8; ++r) {
        const int row = row0 + 16 * hv + rbase + r;
        const int b = row >> 11, s = row & (S_ - 1);
        const float v = acc[j][r];
        if (mode == 0) {
          ((half_t*)outp)[(((size_t)(b * H_ + h) * S_ + s) * DH_) + dh] = (half_t)v;
        } else if (mode == 1) {
          ((half_t*)outp)[(((size_t)(b * H_ + h) * DH_ + dh) * S_) + s] = (half_t)v;
        } else {
          ((float*)outp)[(size_t)row * D_ + col] = v + bias[col];
        }
      }
    }
  }
}

// ---------------------------------------------------------------------------
// Block-sparse flash attention. Grid: (16, B*H=32), block: 256 = 8 waves.
// Double-buffered LDS K/V tiles with async global->LDS copies: while block t
// is consumed by WMMAs, block t+1's copies are in flight into the other buffer.
// Mask row is uniform per workgroup -> all branches/barriers are uniform.
// ---------------------------------------------------------------------------
__global__ __launch_bounds__(256) void attn_blocksparse(
    const half_t* __restrict__ Qh,   // [B,H,S,DH]
    const half_t* __restrict__ Kh,   // [B,H,S,DH]
    const half_t* __restrict__ Vt,   // [B,H,DH,S]
    const unsigned char* __restrict__ mask,  // [NB,NB] bool
    half_t* __restrict__ AOh) {      // [B*S, D] f16
  __shared__ half_t Kl[2][BLK_ * DH_];   // [buf][kcol][dh]  2 x 16 KB
  __shared__ half_t Vl[2][DH_ * BLK_];   // [buf][dh][kcol]  2 x 16 KB
  __shared__ half_t Pl[8][16 * 128];     // per-wave P       32 KB

  const int tid  = threadIdx.x;
  const int wave = tid >> 5;
  const int lane = tid & 31;
  const int qblk = blockIdx.x;
  const int bh   = blockIdx.y;
  const int b    = bh >> 4;
  const int h    = bh & 15;
  const int q0   = qblk * BLK_ + wave * 16;

  const half_t* Qb = Qh + (size_t)bh * S_ * DH_;
  const half_t* Kb = Kh + (size_t)bh * S_ * DH_;
  const half_t* Vb = Vt + (size_t)bh * DH_ * S_;

  // kept-block bitmask (uniform across WG; diagonal guarantees non-empty)
  unsigned kept = 0;
  for (int kb = 0; kb < NB_; ++kb)
    kept |= (mask[qblk * NB_ + kb] ? 1u : 0u) << kb;

  // 8 async 16B copies per thread: 4 for K tile (contiguous 16KB), 4 for V^T
  auto stage_block = [&](int kb, int buf) {
    const int kcol0 = kb * BLK_;
    const half_t* Kg = Kb + (size_t)kcol0 * DH_;
#pragma unroll
    for (int i = 0; i < 4; ++i) {
      const int off = (tid + i * 256) * 8;            // halfs, 16B chunks
      copy16_g2l(&Kl[buf][off], Kg + off);
    }
#pragma unroll
    for (int i = 0; i < 4; ++i) {
      const int t   = tid + i * 256;                  // 0..1023
      const int dh  = t >> 4;                         // 0..63
      const int seg = (t & 15) * 8;                   // halfs within row
      copy16_g2l(&Vl[buf][dh * BLK_ + seg],
                 Vb + (size_t)dh * S_ + kcol0 + seg);
    }
  };

  v16h qa0 = load_a16x32(Qb, DH_, q0, 0);
  v16h qa1 = load_a16x32(Qb, DH_, q0, 32);

  v8f   o[4] = {};
  float mrow[8], lrow[8];
#pragma unroll
  for (int r = 0; r < 8; ++r) { mrow[r] = -1e30f; lrow[r] = 0.0f; }
  const float scale = 0.125f;  // 1/sqrt(DH)

  // ---- pipelined loop over kept blocks ----
  unsigned rem = kept;
  int buf = 0;
  stage_block(__ffs((int)rem) - 1, 0);     // prologue: first kept block

  while (rem) {
    const int kb = __ffs((int)rem) - 1;
    rem &= rem - 1;
    const bool have_next = (rem != 0);
    if (have_next) stage_block(__ffs((int)rem) - 1, buf ^ 1);  // prefetch

    // release current block's (older) copies; next block's may stay in flight
    if (have_next) async_wait_keep8(); else async_wait_all();
    __syncthreads();                       // cross-wave visibility

    const int kcol0 = kb * BLK_;
    const half_t* Kc = &Kl[buf][0];
    const half_t* Vc = &Vl[buf][0];

    // ---- scores S (16x128) = Q @ K^T : 16 WMMA ----
    v8f sc[8];
#pragma unroll
    for (int j = 0; j < 8; ++j) {
      v8f c = {};
      c = wmma_f16(qa0, load_b32x16(Kc, DH_, 16 * j, 0),  c);
      c = wmma_f16(qa1, load_b32x16(Kc, DH_, 16 * j, 32), c);
      sc[j] = c;
    }

    // ---- online softmax ----
#pragma unroll
    for (int r = 0; r < 8; ++r) {
      float mx = -1e30f;
#pragma unroll
      for (int j = 0; j < 8; ++j) mx = fmaxf(mx, sc[j][r]);
#pragma unroll
      for (int d = 1; d < 16; d <<= 1) mx = fmaxf(mx, __shfl_xor(mx, d, 32));
      mx *= scale;
      const float mnew = fmaxf(mrow[r], mx);
      const float corr = __expf(mrow[r] - mnew);
#pragma unroll
      for (int f = 0; f < 4; ++f) o[f][r] *= corr;
      float rsum = 0.0f;
#pragma unroll
      for (int j = 0; j < 8; ++j) {
        const float p = __expf(sc[j][r] * scale - mnew);
        sc[j][r] = p;
        rsum += p;
      }
#pragma unroll
      for (int d = 1; d < 16; d <<= 1) rsum += __shfl_xor(rsum, d, 32);
      lrow[r] = lrow[r] * corr + rsum;
      mrow[r] = mnew;
    }

    // ---- stage P (16x128 f16) into this wave's LDS tile ----
    half_t* P = &Pl[wave][0];
#pragma unroll
    for (int j = 0; j < 8; ++j) {
      const int col = 16 * j + (lane & 15);
#pragma unroll
      for (int r = 0; r < 8; ++r) {
        const int row = ((lane >> 4) << 3) + r;
        P[row * 128 + col] = (half_t)sc[j][r];
      }
    }

    // ---- O (16x64) += P @ V : 16 WMMA ----
#pragma unroll
    for (int kk = 0; kk < 4; ++kk) {
      v16h pa;
      {
        const int m   = lane & 15;
        const int kbp = kk * 32 + ((lane >> 4) << 3);
        v8h lo = *reinterpret_cast<const v8h*>(&Pl[wave][m * 128 + kbp]);
        v8h hi = *reinterpret_cast<const v8h*>(&Pl[wave][m * 128 + kbp + 16]);
#pragma unroll
        for (int i = 0; i < 8; ++i) { pa[i] = lo[i]; pa[8 + i] = hi[i]; }
      }
#pragma unroll
      for (int f = 0; f < 4; ++f) {
        o[f] = wmma_f16(pa, load_b32x16(Vc, BLK_, 16 * f, kk * 32), o[f]);
      }
    }
    __syncthreads();   // all waves done reading buf before it is re-staged
    buf ^= 1;
    (void)kcol0;
  }

  // ---- normalize, merge heads, store ----
#pragma unroll
  for (int r = 0; r < 8; ++r) {
    const float inv = 1.0f / lrow[r];
#pragma unroll
    for (int f = 0; f < 4; ++f) o[f][r] *= inv;
  }
#pragma unroll
  for (int f = 0; f < 4; ++f) {
    const int dh = 16 * f + (lane & 15);
#pragma unroll
    for (int r = 0; r < 8; ++r) {
      const int s = q0 + ((lane >> 4) << 3) + r;
      AOh[(size_t)(b * S_ + s) * D_ + h * DH_ + dh] = (half_t)o[f][r];
    }
  }
}

// ---------------------------------------------------------------------------
extern "C" void kernel_launch(void* const* d_in, const int* in_sizes, int n_in,
                              void* d_out, int out_size, void* d_ws,
                              size_t ws_size, hipStream_t stream) {
  (void)in_sizes; (void)n_in; (void)out_size; (void)ws_size;
  const float* x  = (const float*)d_in[0];
  const float* Wq = (const float*)d_in[1];
  const float* Wk = (const float*)d_in[2];
  const float* Wv = (const float*)d_in[3];
  const float* Wo = (const float*)d_in[4];
  const float* bo = (const float*)d_in[5];
  const unsigned char* mask = (const unsigned char*)d_in[6];

  half_t* ws  = (half_t*)d_ws;
  half_t* xh  = ws;                                // 4096*1024
  half_t* wqh = xh  + (size_t)4096 * 1024;
  half_t* wkh = wqh + (size_t)1024 * 1024;
  half_t* wvh = wkh + (size_t)1024 * 1024;
  half_t* woh = wvh + (size_t)1024 * 1024;
  half_t* qh  = woh + (size_t)1024 * 1024;         // 4M each
  half_t* kh  = qh  + (size_t)4 * 1024 * 1024;
  half_t* vt  = kh  + (size_t)4 * 1024 * 1024;
  half_t* aoh = vt  + (size_t)4 * 1024 * 1024;

  const int nX = 4096 * 1024, nW = 1024 * 1024;
  cvt_f32_to_f16<<<(nX + 255) / 256, 256, 0, stream>>>(x,  xh,  nX);
  cvt_f32_to_f16<<<(nW + 255) / 256, 256, 0, stream>>>(Wq, wqh, nW);
  cvt_f32_to_f16<<<(nW + 255) / 256, 256, 0, stream>>>(Wk, wkh, nW);
  cvt_f32_to_f16<<<(nW + 255) / 256, 256, 0, stream>>>(Wv, wvh, nW);
  cvt_f32_to_f16<<<(nW + 255) / 256, 256, 0, stream>>>(Wo, woh, nW);

  const dim3 gG(16, 16), blk(256);
  gemm_xwT<<<gG, blk, 0, stream>>>(xh, wqh, qh, nullptr, 0);   // Q
  gemm_xwT<<<gG, blk, 0, stream>>>(xh, wkh, kh, nullptr, 0);   // K
  gemm_xwT<<<gG, blk, 0, stream>>>(xh, wvh, vt, nullptr, 1);   // V^T

  attn_blocksparse<<<dim3(16, 32), blk, 0, stream>>>(qh, kh, vt, mask, aoh);

  gemm_xwT<<<gG, blk, 0, stream>>>(aoh, woh, d_out, bo, 2);    // out + bias
}
